// SelfAttention_37503654428807
// MI455X (gfx1250) — matrix operand order repaired and checked by reading, not compiled, not run on hardware
//
#include <hip/hip_runtime.h>
#include <math.h>

// ---------------------------------------------------------------------------
// Self-attention (B=8, C=256, N=4096):
//   f,g = Wf/Wg x + b   [B,32,N] bf16 (ws);  h = Wh x + bh [B,256,N] bf16 (ws)
//   s = f^T g  -> fp32 straight into d_out attn slot; softmax in place
//   out = gamma * (h @ attn^T) + x  (fp32)
// GEMMs: v_wmma_f32_16x16x32_bf16.  Dominant GEMM: 128x128 WG tile, TDM
// tensor_load_to_lds double-buffered A tiles, reg-double-buffered B tiles,
// 8 WMMAs/wave/K-step with reuse_a operand-cache hints.
// ---------------------------------------------------------------------------

typedef __attribute__((ext_vector_type(16))) __bf16 v16bf;
typedef __attribute__((ext_vector_type(8)))  __bf16 v8bf;
typedef __attribute__((ext_vector_type(8)))  float  v8f;
typedef __attribute__((ext_vector_type(4)))  float  v4f;
typedef unsigned short u16;
typedef __attribute__((ext_vector_type(4)))  u16 v4us;
typedef __attribute__((ext_vector_type(8)))  u16 v8us;
typedef __attribute__((ext_vector_type(4)))  unsigned int v4u;
typedef __attribute__((ext_vector_type(4)))  int v4i;
typedef __attribute__((ext_vector_type(8)))  int v8i;

#define BB   8
#define CC   256
#define C8   32
#define NN   4096
#define OUT_ELEMS ((size_t)BB * CC * NN)

#if __has_builtin(__builtin_amdgcn_tensor_load_to_lds)
#define HAVE_TDM 1
#if __has_include(<hip/amd_detail/amd_gfx1250_TDM.h>)
#define TDM_SIX_ARGS 1
#endif
#else
#define HAVE_TDM 0
#endif

// ---- helpers ---------------------------------------------------------------

static __device__ __forceinline__ u16 f2bf(float f) {
    unsigned u = __float_as_uint(f);
    u += 0x7FFFu + ((u >> 16) & 1u);
    return (u16)(u >> 16);
}

static __device__ __forceinline__ v4us f2bf4(v4f f) {
    v4us r;
#pragma unroll
    for (int i = 0; i < 4; ++i) r[i] = f2bf(f[i]);
    return r;
}

// A fragment 16x32 (MxK) from row-major LDS tile [m][k], ld = 32.
static __device__ __forceinline__ v16bf frag_a(const u16* lds, int row0) {
    const int lane = threadIdx.x & 31;
    const int half = lane >> 4, lr = lane & 15;
    const u16* p = lds + (row0 + lr) * 32 + 16 * half;
    v8bf lo = *(const v8bf*)(p);
    v8bf hi = *(const v8bf*)(p + 8);
    return __builtin_shufflevector(lo, hi, 0, 1, 2, 3, 4, 5, 6, 7,
                                           8, 9, 10, 11, 12, 13, 14, 15);
}

// B fragment 32x16 (KxN) from TRANSPOSED LDS tile [n][k], ld = 32.
static __device__ __forceinline__ v16bf frag_bT(const u16* lds, int col0) {
    const int lane = threadIdx.x & 31;
    const int half = lane >> 4, lr = lane & 15;
    return *(const v16bf*)(lds + (col0 + lr) * 32 + 16 * half);
}

static __device__ __forceinline__ v8f wmma_bf16(v16bf a, v16bf b, v8f c) {
    return __builtin_amdgcn_wmma_f32_16x16x32_bf16(false, a, false, b, (short)0, c,
                                                   false, false);
}

static __device__ __forceinline__ v8f wmma_bf16_ra(v16bf a, v16bf b, v8f c) {
    // identical instruction as previous WMMA, same A operand -> reuse_a legal
    return __builtin_amdgcn_wmma_f32_16x16x32_bf16(false, a, false, b, (short)0, c,
                                                   true, false);
}

#if HAVE_TDM
static __device__ __forceinline__ unsigned lds_off(const void* p) {
    return (unsigned)(unsigned long long)p;   // LDS flat addr: low 32 bits = offset
}

// 2D bf16 tile load: tile_dim0 = 32 elems (contiguous k), tile_dim1 = rows,
// row stride = stride elems. D# per CDNA5 ISA 8.3/8.4.
static __device__ __forceinline__ void tdm_load_bf16_tile(
    unsigned lds_addr, const void* gptr, unsigned rows, unsigned long long stride) {
    unsigned long long ga = (unsigned long long)gptr;
    unsigned s_lo = (unsigned)stride;
    unsigned s_hi = (unsigned)(stride >> 32) & 0xFFFFu;
    v4u g0;
    g0[0] = 1u;                                   // count=1, user descriptor
    g0[1] = lds_addr;                             // lds_addr (bytes)
    g0[2] = (unsigned)ga;                         // global_addr[31:0]
    g0[3] = ((unsigned)(ga >> 32) & 0x01FFFFFFu) | 0x80000000u; // addr[56:32]|type=2
    v8i g1;
    g1[0] = (int)0x00010000u;                     // data_size=1 (2 bytes), no flags
    g1[1] = (int)((s_lo & 0xFFFFu) << 16);        // tensor_dim0[15:0]
    g1[2] = (int)(((s_lo >> 16) & 0xFFFFu) | (rows << 16)); // dim0 hi | dim1 lo
    g1[3] = (int)((rows >> 16) | (32u << 16));    // dim1 hi | tile_dim0=32
    g1[4] = (int)(rows & 0xFFFFu);                // tile_dim1=rows, tile_dim2=0
    g1[5] = (int)s_lo;                            // tensor_dim0_stride[31:0]
    g1[6] = (int)s_hi;                            // stride[47:32]
    g1[7] = 0;
    v4i gz = {0, 0, 0, 0};
#if defined(TDM_SIX_ARGS)
    v8i gz8 = {0, 0, 0, 0, 0, 0, 0, 0};
    __builtin_amdgcn_tensor_load_to_lds(g0, g1, gz, gz, gz8, 0);
#else
    __builtin_amdgcn_tensor_load_to_lds(g0, g1, gz, gz, 0);
#endif
}
#endif

// ---------------------------------------------------------------------------
// Kernel 1: projection GEMM  Out[M,N] = W[M,K]*X[K,N] + bias (per batch), bf16 out
// ---------------------------------------------------------------------------
__global__ void __launch_bounds__(256)
proj_gemm(const float* __restrict__ X, const float* __restrict__ W,
          const float* __restrict__ bias, u16* __restrict__ Out,
          int M, int K) {
    __shared__ __align__(16) u16 As[32 * 32];   // [m][k]
    __shared__ __align__(16) u16 Bs[64 * 32];   // [n][k] (transposed)

    const int n0 = blockIdx.x * 64, m0 = blockIdx.y * 32, b = blockIdx.z;
    const int tid = threadIdx.x;
    const int wid = tid >> 5, wm = wid >> 2, wn = wid & 3;
    const int lane = tid & 31, half = lane >> 4, lr = lane & 15;

    const float* Xb = X + (size_t)b * K * NN;

    const int mA = tid >> 3, kA = (tid & 7) * 4;
    const int nB = (tid & 15) * 4, kB = tid >> 4;

    v8f acc = {};
    for (int kk = 0; kk < K; kk += 32) {
        __syncthreads();
        v4f wv = *(const v4f*)&W[(size_t)(m0 + mA) * K + kk + kA];
        v4f x0 = *(const v4f*)&Xb[(size_t)(kk + kB) * NN + n0 + nB];
        v4f x1 = *(const v4f*)&Xb[(size_t)(kk + 16 + kB) * NN + n0 + nB];
        *(v4us*)&As[mA * 32 + kA] = f2bf4(wv);
        v4us xb0 = f2bf4(x0), xb1 = f2bf4(x1);
#pragma unroll
        for (int j = 0; j < 4; ++j) {
            Bs[(nB + j) * 32 + kB]      = xb0[j];
            Bs[(nB + j) * 32 + kB + 16] = xb1[j];
        }
        __syncthreads();
        acc = wmma_bf16(frag_a(As, wm * 16), frag_bT(Bs, wn * 16), acc);
    }

#pragma unroll
    for (int r = 0; r < 8; ++r) {
        int m = m0 + wm * 16 + r + 8 * half;
        int n = n0 + wn * 16 + lr;
        Out[(size_t)b * M * NN + (size_t)m * NN + n] = f2bf(acc[r] + bias[m]);
    }
}

// ---------------------------------------------------------------------------
// Kernel 2: score GEMM s[i,j] = sum_k f[k,i] g[k,j], K=32 (one WMMA / tile).
// ---------------------------------------------------------------------------
__global__ void __launch_bounds__(256)
attn_score(const u16* __restrict__ F, const u16* __restrict__ G,
           float* __restrict__ S) {
    __shared__ __align__(16) u16 As[32 * 32];   // [i][k]
    __shared__ __align__(16) u16 Bs[64 * 32];   // [j][k] (transposed)

    const int j0 = blockIdx.x * 64, i0 = blockIdx.y * 32, b = blockIdx.z;
    const int tid = threadIdx.x;
    const int wid = tid >> 5, wm = wid >> 2, wn = wid & 3;
    const int lane = tid & 31, half = lane >> 4, lr = lane & 15;

    const u16* Fb = F + (size_t)b * C8 * NN;
    const u16* Gb = G + (size_t)b * C8 * NN;

    {
        const int mA = (tid & 7) * 4, kA = tid >> 3;
        v4us fa = *(const v4us*)&Fb[(size_t)kA * NN + i0 + mA];
#pragma unroll
        for (int j = 0; j < 4; ++j) As[(mA + j) * 32 + kA] = fa[j];
    }
    {
        const int nB = (tid & 15) * 4, kB = tid >> 4;
        v4us g0v = *(const v4us*)&Gb[(size_t)kB * NN + j0 + nB];
        v4us g1v = *(const v4us*)&Gb[(size_t)(kB + 16) * NN + j0 + nB];
#pragma unroll
        for (int j = 0; j < 4; ++j) {
            Bs[(nB + j) * 32 + kB]      = g0v[j];
            Bs[(nB + j) * 32 + kB + 16] = g1v[j];
        }
    }
    __syncthreads();

    v8f acc = {};
    acc = wmma_bf16(frag_a(As, wm * 16), frag_bT(Bs, wn * 16), acc);

#pragma unroll
    for (int r = 0; r < 8; ++r) {
        int i = i0 + wm * 16 + r + 8 * half;
        int j = j0 + wn * 16 + lr;
        S[(size_t)b * NN * NN + (size_t)i * NN + j] = acc[r];
    }
}

// ---------------------------------------------------------------------------
// Kernel 3: row softmax in place; 256 thr per 4096-elem row; b128 loads.
// ---------------------------------------------------------------------------
__global__ void __launch_bounds__(256)
softmax_rows(float* __restrict__ S) {
    __shared__ float red[256];
    const int tid = threadIdx.x;
    float* row = S + (size_t)blockIdx.x * NN;

    v4f v[4];
    float mx = -INFINITY;
#pragma unroll
    for (int e = 0; e < 4; ++e) {
        v[e] = *(const v4f*)&row[tid * 4 + e * 1024];
#pragma unroll
        for (int j = 0; j < 4; ++j) mx = fmaxf(mx, v[e][j]);
    }
    red[tid] = mx; __syncthreads();
    for (int s = 128; s > 0; s >>= 1) {
        if (tid < s) red[tid] = fmaxf(red[tid], red[tid + s]);
        __syncthreads();
    }
    mx = red[0]; __syncthreads();

    float sum = 0.f;
#pragma unroll
    for (int e = 0; e < 4; ++e)
#pragma unroll
        for (int j = 0; j < 4; ++j) {
            v[e][j] = __expf(v[e][j] - mx);
            sum += v[e][j];
        }
    red[tid] = sum; __syncthreads();
    for (int s = 128; s > 0; s >>= 1) {
        if (tid < s) red[tid] += red[tid + s];
        __syncthreads();
    }
    const float inv = 1.f / red[0];
#pragma unroll
    for (int e = 0; e < 4; ++e) {
#pragma unroll
        for (int j = 0; j < 4; ++j) v[e][j] *= inv;
        *(v4f*)&row[tid * 4 + e * 1024] = v[e];
    }
}

// ---------------------------------------------------------------------------
// Kernel 4: out GEMM out[c,i] = sum_j h[c,j] attn[i,j]; M=256,N=4096,K=4096.
// 128x128 WG tile, 8 waves; wave w owns rows [w*16, w*16+16) x all 128 cols:
// 1 A frag, 8 B frags, 8 WMMAs per K-step (reuse_a set on WMMAs 2..8).
// A tiles: TDM tensor_load_to_lds, double buffered, issued one step ahead.
// B tiles: fp32 attn, register double buffered, fp32->bf16 on LDS store.
// ---------------------------------------------------------------------------
#define MT 128
#define NT 128

__global__ void __launch_bounds__(256)
attn_out(const u16* __restrict__ H, const float* __restrict__ Attn,
         const float* __restrict__ X, const float* __restrict__ gamma,
         float* __restrict__ Out) {
    __shared__ __align__(16) u16 As[2][MT * 32];  // [m][k], double buffer (16KB)
    __shared__ __align__(16) u16 Bs[NT * 32];     // [n][k] transposed (8KB)

    const int n0 = blockIdx.x * NT, m0 = blockIdx.y * MT, b = blockIdx.z;
    const int tid = threadIdx.x;
    const int wid = tid >> 5;
    const int lane = tid & 31, half = lane >> 4, lr = lane & 15;

    const u16*   Hb = H + (size_t)b * CC * NN + (size_t)m0 * NN;
    const float* Ab = Attn + (size_t)b * NN * NN;

    const int nrow = tid >> 1;          // 0..127
    const int kb   = (tid & 1) * 16;    // 0 or 16
    const float* bp = Ab + (size_t)(n0 + nrow) * NN + kb;

    v4f rb[4];
#pragma unroll
    for (int j = 0; j < 4; ++j) rb[j] = *(const v4f*)(bp + j * 4);

#if HAVE_TDM
    if (tid < 32)
        tdm_load_bf16_tile(lds_off(&As[0][0]), Hb, MT, NN);
#endif

    v8f acc[8];
#pragma unroll
    for (int t = 0; t < 8; ++t) acc[t] = (v8f){};

    for (int kk = 0; kk < NN; kk += 32) {
        const int buf = (kk >> 5) & 1;
        __syncthreads();
#pragma unroll
        for (int j = 0; j < 4; ++j)
            *(v4us*)&Bs[nrow * 32 + kb + j * 4] = f2bf4(rb[j]);
        if (kk + 32 < NN) {             // issue next B loads early
#pragma unroll
            for (int j = 0; j < 4; ++j)
                rb[j] = *(const v4f*)(bp + kk + 32 + j * 4);
        }
#if HAVE_TDM
        if (tid < 32) {
            __builtin_amdgcn_s_wait_tensorcnt(0);       // As[buf] DMA complete
            if (kk + 32 < NN)
                tdm_load_bf16_tile(lds_off(&As[buf ^ 1][0]), Hb + kk + 32, MT, NN);
        }
        const u16* Acur = &As[buf][0];
#else
        *(v8us*)&As[0][nrow * 32 + kb]     = *(const v8us*)&Hb[(size_t)nrow * NN + kk + kb];
        *(v8us*)&As[0][nrow * 32 + kb + 8] = *(const v8us*)&Hb[(size_t)nrow * NN + kk + kb + 8];
        const u16* Acur = &As[0][0];
#endif
        __syncthreads();
        v16bf a = frag_a(Acur, wid * 16);
        // first WMMA: no reuse hint; subsequent 7 share identical A operand
        acc[0] = wmma_bf16(a, frag_bT(Bs, 0), acc[0]);
#pragma unroll
        for (int t = 1; t < 8; ++t)
            acc[t] = wmma_bf16_ra(a, frag_bT(Bs, t * 16), acc[t]);
    }

    const float gm = gamma[0];
#pragma unroll
    for (int t = 0; t < 8; ++t) {
#pragma unroll
        for (int r = 0; r < 8; ++r) {
            int m = m0 + wid * 16 + r + 8 * half;
            int n = n0 + t * 16 + lr;
            size_t idx = (size_t)b * CC * NN + (size_t)m * NN + n;
            Out[idx] = gm * acc[t][r] + X[idx];
        }
    }
}

// ---------------------------------------------------------------------------

extern "C" void kernel_launch(void* const* d_in, const int* in_sizes, int n_in,
                              void* d_out, int out_size, void* d_ws, size_t ws_size,
                              hipStream_t stream) {
    (void)in_sizes; (void)n_in; (void)out_size; (void)ws_size;

    const float* x     = (const float*)d_in[0];
    const float* Wf    = (const float*)d_in[1];
    const float* bf    = (const float*)d_in[2];
    const float* Wg    = (const float*)d_in[3];
    const float* bg    = (const float*)d_in[4];
    const float* Wh    = (const float*)d_in[5];
    const float* bh    = (const float*)d_in[6];
    const float* gamma = (const float*)d_in[7];

    float* out  = (float*)d_out;
    float* attn = out + OUT_ELEMS;

    u16* F = (u16*)d_ws;
    u16* G = F + (size_t)BB * C8 * NN;
    u16* H = G + (size_t)BB * C8 * NN;

    proj_gemm<<<dim3(NN / 64, C8 / 32, BB), 256, 0, stream>>>(x, Wf, bf, F, C8, CC);
    proj_gemm<<<dim3(NN / 64, C8 / 32, BB), 256, 0, stream>>>(x, Wg, bg, G, C8, CC);
    proj_gemm<<<dim3(NN / 64, CC / 32, BB), 256, 0, stream>>>(x, Wh, bh, H, CC, CC);

    attn_score<<<dim3(NN / 64, NN / 32, BB), 256, 0, stream>>>(F, G, attn);
    softmax_rows<<<dim3(BB * NN), 256, 0, stream>>>(attn);
    attn_out<<<dim3(NN / NT, CC / MT, BB), 256, 0, stream>>>(H, attn, x, gamma, out);
}
